// GaussianWindow_58093727645935
// MI455X (gfx1250) — matrix operand order repaired and checked by reading, not compiled, not run on hardware
//
#include <hip/hip_runtime.h>
#include <hip/hip_bf16.h>

typedef _Float16 half8 __attribute__((ext_vector_type(8)));
typedef _Float16 v16h  __attribute__((ext_vector_type(16)));
typedef float    v8f   __attribute__((ext_vector_type(8)));
typedef float    v4f   __attribute__((ext_vector_type(4)));

#define BB 512
#define VV 1024
#define KK 64
#define CC 2048
#define AA 128

// ---------------------------------------------------------------------------
// Kernel 0a: convert x (B,V) f32 -> f16
// ---------------------------------------------------------------------------
__global__ __launch_bounds__(256) void gw_cvt_x(const float* __restrict__ x,
                                                _Float16* __restrict__ xh, int n) {
    int i = blockIdx.x * 256 + threadIdx.x;
    if (i < n) xh[i] = (_Float16)x[i];
}

// ---------------------------------------------------------------------------
// Kernel 0b: build Wt[ng][v] = W_mat[v][k] in f16, ng = mat*64 + k, mat in
// {Wa,Wb,Wk}. Row-major Wt means WMMA B-fragment loads are contiguous.
// ---------------------------------------------------------------------------
__global__ __launch_bounds__(256) void gw_cvt_w(const float* __restrict__ Wa,
                                                const float* __restrict__ Wb,
                                                const float* __restrict__ Wk,
                                                _Float16* __restrict__ wt) {
    int i = blockIdx.x * 256 + threadIdx.x;      // 0 .. 192*1024-1
    if (i >= 3 * KK * VV) return;
    int ng = i >> 10;            // output feature row 0..191
    int v  = i & (VV - 1);       // 0..1023
    int mat = ng >> 6;
    int k   = ng & 63;
    const float* W = (mat == 0) ? Wa : (mat == 1) ? Wb : Wk;
    wt[i] = (_Float16)W[v * KK + k];
}

// ---------------------------------------------------------------------------
// Kernel 1: WMMA GEMM  s = xs @ [Wa|Wb|Wk]  (M=512, N=192, K=1024), fused
// activation epilogue:
//   mat 0 -> log_alpha = s + ba          (keep pre-exp for fused phi)
//   mat 1 -> beta      = exp(s + bb)
//   mat 2 -> new_kappa = prev_kappa + exp(s + bk)   (written to d_out)
// One wave per 16x16 tile. grid = (B/16, 192/16).
// ---------------------------------------------------------------------------
__global__ __launch_bounds__(32) void gw_gemm_act(
    const _Float16* __restrict__ xh, const _Float16* __restrict__ wt,
    const float* __restrict__ ba, const float* __restrict__ bb,
    const float* __restrict__ bk, const float* __restrict__ prev_kappa,
    float* __restrict__ la_out, float* __restrict__ beta_out,
    float* __restrict__ kappa_out) {

    const int lane = threadIdx.x;        // 0..31
    const int m0   = blockIdx.x * 16;
    const int n0   = blockIdx.y * 16;
    const int rsel = lane >> 4;          // 0: lanes 0-15, 1: lanes 16-31
    const int lmod = lane & 15;

    // A fragment (16x32, row-major source): lane<16 row=lmod K{0..7,16..23},
    // lane>=16 row=lmod K{8..15,24..31}.
    const _Float16* arow = xh + (size_t)(m0 + lmod) * VV + rsel * 8;
    // B fragment (32x16): lane<16 col=lmod K{0..7,16..23}, lane>=16 col=lmod K{8..15,24..31}.
    // Column n of B == row n of Wt (contiguous).
    const _Float16* brow = wt + (size_t)(n0 + lmod) * VV + rsel * 8;

    v8f acc = {};
    for (int k0 = 0; k0 < VV; k0 += 32) {
        half8 a_lo = *(const half8*)(arow + k0);        // 16B aligned
        half8 a_hi = *(const half8*)(arow + k0 + 16);
        half8 b_lo = *(const half8*)(brow + k0);
        half8 b_hi = *(const half8*)(brow + k0 + 16);
        v16h A, Bm;
#pragma unroll
        for (int i = 0; i < 8; ++i) {
            A[i] = a_lo[i];  A[i + 8] = a_hi[i];
            Bm[i] = b_lo[i]; Bm[i + 8] = b_hi[i];
        }
        acc = __builtin_amdgcn_wmma_f32_16x16x32_f16(
            /*neg_a=*/false, A, /*neg_b=*/false, Bm,
            /*c_mod=*/(short)0, acc, /*reuse_a=*/false, /*reuse_b=*/false);
    }

    // D layout: VGPR r -> row m0+r (lanes 0-15) / m0+8+r (lanes 16-31), col lmod
    const int ng  = n0 + lmod;
    const int mat = ng >> 6;
    const int k   = ng & 63;
#pragma unroll
    for (int r = 0; r < 8; ++r) {
        int row = m0 + r + rsel * 8;
        int idx = row * KK + k;
        float s = acc[r];
        if (mat == 0) {
            la_out[idx] = s + ba[k];
        } else if (mat == 1) {
            beta_out[idx] = __expf(s + bb[k]);
        } else {
            kappa_out[idx] = prev_kappa[idx] + __expf(s + bk[k]);
        }
    }
}

// ---------------------------------------------------------------------------
// Kernel 2: per-batch fused phi + window. Block = 256 threads, one block per b.
//   phi[c]   = sum_k exp(log_alpha[k] - beta[k]*(kappa[k]-c)^2)   -> LDS + out
//   window[a]= sum_c phi[c] * text[b,c,a]                          -> out
// text (512 MB) is streamed once with non-temporal float4 loads.
// ---------------------------------------------------------------------------
__global__ __launch_bounds__(256) void gw_phi_window(
    const float* __restrict__ text, const float* __restrict__ la,
    const float* __restrict__ beta, const float* __restrict__ kappa,
    float* __restrict__ phi_out, float* __restrict__ win_out) {

    __shared__ float s_la[KK], s_b[KK], s_k[KK];
    __shared__ float s_phi[CC];
    __shared__ float s_win[8][AA];

    const int b = blockIdx.x;
    const int t = threadIdx.x;

    if (t < KK) {
        s_la[t] = la[b * KK + t];
        s_b[t]  = beta[b * KK + t];
        s_k[t]  = kappa[b * KK + t];
    }
    __syncthreads();

    // ---- phi ----
    for (int c = t; c < CC; c += 256) {
        float fc  = (float)c;
        float acc = 0.0f;
#pragma unroll 16
        for (int k = 0; k < KK; ++k) {
            float d = s_k[k] - fc;
            acc += __expf(s_la[k] - s_b[k] * d * d);
        }
        s_phi[c] = acc;
        phi_out[(size_t)b * CC + c] = acc;
    }
    __syncthreads();

    // ---- window: lane a4 covers a = 4*a4..4*a4+3, cg covers c = cg mod 8 ----
    const int a4 = t & 31;
    const int cg = t >> 5;
    const float* tb = text + (size_t)b * CC * AA;
    v4f acc4 = {};
#pragma unroll 4
    for (int c = cg; c < CC; c += 8) {
        const v4f* p = (const v4f*)(tb + c * AA + a4 * 4);
        v4f v = __builtin_nontemporal_load(p);       // read-once stream
        float ph = s_phi[c];
        v4f pv = {ph, ph, ph, ph};
        acc4 += pv * v;
    }
#pragma unroll
    for (int i = 0; i < 4; ++i) s_win[cg][a4 * 4 + i] = acc4[i];
    __syncthreads();

    if (t < AA) {
        float s = 0.0f;
#pragma unroll
        for (int g = 0; g < 8; ++g) s += s_win[g][t];
        win_out[(size_t)b * AA + t] = s;
    }
}

// ---------------------------------------------------------------------------
// launch
// ---------------------------------------------------------------------------
extern "C" void kernel_launch(void* const* d_in, const int* in_sizes, int n_in,
                              void* d_out, int out_size, void* d_ws, size_t ws_size,
                              hipStream_t stream) {
    const float* x          = (const float*)d_in[0];   // (B,1,V)
    const float* text       = (const float*)d_in[1];   // (B,C,A)
    const float* prev_kappa = (const float*)d_in[2];   // (B,K)
    const float* Wa         = (const float*)d_in[3];
    const float* ba         = (const float*)d_in[4];
    const float* Wb         = (const float*)d_in[5];
    const float* bb         = (const float*)d_in[6];
    const float* Wk         = (const float*)d_in[7];
    const float* bk         = (const float*)d_in[8];

    float* out       = (float*)d_out;
    float* phi_out   = out;                        // B*C
    float* kappa_out = out + (size_t)BB * CC;      // B*K
    float* win_out   = kappa_out + (size_t)BB * KK;// B*A

    char* ws = (char*)d_ws;
    _Float16* xh = (_Float16*)ws;                               // 1 MB
    _Float16* wt = (_Float16*)(ws + (size_t)BB * VV * 2);       // 384 KB
    float* la    = (float*)(ws + (size_t)BB * VV * 2 + (size_t)3 * KK * VV * 2);
    float* beta  = la + (size_t)BB * KK;                        // 128 KB each

    gw_cvt_x<<<(BB * VV + 255) / 256, 256, 0, stream>>>(x, xh, BB * VV);
    gw_cvt_w<<<(3 * KK * VV + 255) / 256, 256, 0, stream>>>(Wa, Wb, Wk, wt);
    gw_gemm_act<<<dim3(BB / 16, (3 * KK) / 16), 32, 0, stream>>>(
        xh, wt, ba, bb, bk, prev_kappa, la, beta, kappa_out);
    gw_phi_window<<<BB, 256, 0, stream>>>(text, la, beta, kappa_out,
                                          phi_out, win_out);
}
